// MhcModule_41798621724799
// MI455X (gfx1250) — matrix operand order repaired and checked by reading, not compiled, not run on hardware
//
#include <hip/hip_runtime.h>
#include <math.h>

#define NS 4
#define EPS 1e-6f
#define SINK_ITERS 20
#define S_TOT 4096
#define D_TOT 4096
#define TILE_S 16
#define NWAVES 8
#define KSPAN (D_TOT / NWAVES) /* 512 per wave */

typedef __attribute__((ext_vector_type(16))) _Float16 v16h;
typedef __attribute__((ext_vector_type(8)))  float    v8f;

__device__ __forceinline__ float sigmoidf_(float z) {
    return 1.0f / (1.0f + __expf(-z));
}

// Load 8 contiguous floats, scale by gamma*rs, convert to f16.
__device__ __forceinline__ void load8n(const float* xp, const float* gp, float rs,
                                       _Float16* dst) {
    float4 a0 = *(const float4*)(xp);
    float4 a1 = *(const float4*)(xp + 4);
    float4 g0 = *(const float4*)(gp);
    float4 g1 = *(const float4*)(gp + 4);
    dst[0] = (_Float16)(a0.x * g0.x * rs);
    dst[1] = (_Float16)(a0.y * g0.y * rs);
    dst[2] = (_Float16)(a0.z * g0.z * rs);
    dst[3] = (_Float16)(a0.w * g0.w * rs);
    dst[4] = (_Float16)(a1.x * g1.x * rs);
    dst[5] = (_Float16)(a1.y * g1.y * rs);
    dst[6] = (_Float16)(a1.z * g1.z * rs);
    dst[7] = (_Float16)(a1.w * g1.w * rs);
}

// Load 16 contiguous floats, scale by sc, convert to f16. sc is 0 or 1 to
// mask padded N columns without EXEC divergence in the WMMA K-loop.
__device__ __forceinline__ void load16ps(const float* pp, float sc, _Float16* dst) {
#pragma unroll
    for (int q = 0; q < 4; ++q) {
        float4 v = *(const float4*)(pp + 4 * q);
        dst[4 * q + 0] = (_Float16)(v.x * sc);
        dst[4 * q + 1] = (_Float16)(v.y * sc);
        dst[4 * q + 2] = (_Float16)(v.z * sc);
        dst[4 * q + 3] = (_Float16)(v.w * sc);
    }
}

__global__ __launch_bounds__(256) void mhc_fused_kernel(
    const float* __restrict__ x,       // [S, D]
    const float* __restrict__ phi,     // [24, D]
    const float* __restrict__ alpha,   // [3]
    const float* __restrict__ beta,    // [24]
    const float* __restrict__ gamma_,  // [D]
    float* __restrict__ out)           // [S, D]
{
    __shared__ float s_part[NWAVES][2][8][32]; // partial WMMA C tiles (16 KB)
    __shared__ float s_red[TILE_S][16];        // sumsq partials
    __shared__ float s_rs[TILE_S];             // rsqrt(mean(x^2)+eps)
    __shared__ float s_W[TILE_S][32];          // projection result (24 valid)
    __shared__ float s_M[TILE_S][16];          // per-token 4x4 mixing matrix

    const int tid  = threadIdx.x;
    const int wave = tid >> 5;
    const int lane = tid & 31;
    const int s0   = blockIdx.x * TILE_S;

    // ---------------- Phase 1: RMS sums (coalesced, 16 threads per token) ----
    {
        const int m   = tid >> 4;  // token 0..15
        const int sub = tid & 15;
        const float4* xr = (const float4*)(x + (size_t)(s0 + m) * D_TOT);
        float acc = 0.0f;
#pragma unroll 4
        for (int j = 0; j < 64; ++j) {
            float4 v = xr[sub + j * 16];
            acc += v.x * v.x + v.y * v.y + v.z * v.z + v.w * v.w;
        }
        s_red[m][sub] = acc;
    }
    __syncthreads();
    if (tid < TILE_S) {
        float s = 0.0f;
#pragma unroll
        for (int j = 0; j < 16; ++j) s += s_red[tid][j];
        s_rs[tid] = rsqrtf(s * (1.0f / (float)D_TOT) + EPS);
    }
    __syncthreads();

    // ---------------- Phase 2: W = x_normed @ phi^T via v_wmma_f32_16x16x32_f16
    // Each wave covers a 512-wide K slice; partial C reduced through LDS.
    {
        const int   m   = lane & 15;   // token row of A
        const int   hi  = lane >> 4;   // lane half selects K sub-block
        const float rs  = s_rs[m];
        const float* xrow  = x + (size_t)(s0 + m) * D_TOT;
        const int   n0  = lane & 15;   // output column within N-tile
        const float* prow0 = phi + (size_t)n0 * D_TOT;   // n = n0 (<16, valid)
        // N-tile 1: n = 16+n0; rows >= 24 are padding. Clamp pointer to a
        // valid row and zero the contribution via sc1 (keeps loop uniform,
        // no EXEC save/restore around the WMMA).
        const int   n1   = (16 + n0 < 24) ? (16 + n0) : 23;
        const float sc1  = (16 + n0 < 24) ? 1.0f : 0.0f;
        const float* prow1 = phi + (size_t)n1 * D_TOT;

        v8f c0 = {};
        v8f c1 = {};
        const int kwb = wave * KSPAN;

        for (int kb = 0; kb < KSPAN / 32; ++kb) {
            const int kbase = kwb + kb * 32;

            // ---- A fragment: token m, k = kbase + hi*16 + {0..15}
            _Float16 ta[16];
            const int k0 = kbase + hi * 16;
            load8n(xrow + k0,      gamma_ + k0,      rs, ta);
            load8n(xrow + k0 + 8,  gamma_ + k0 + 8,  rs, ta + 8);
            v16h a;
#pragma unroll
            for (int e = 0; e < 16; ++e) a[e] = ta[e];

            // ---- B fragments: col n, k = kbase + hi*16 + e (16 contiguous)
            const int kB = kbase + hi * 16;
            _Float16 tb[16];
            load16ps(prow0 + kB, 1.0f, tb);
            v16h b0;
#pragma unroll
            for (int e = 0; e < 16; ++e) b0[e] = tb[e];

            load16ps(prow1 + kB, sc1, tb);
            v16h b1;
#pragma unroll
            for (int e = 0; e < 16; ++e) b1[e] = tb[e];

            // D = A*B + C   (f32 accumulate)
            c0 = __builtin_amdgcn_wmma_f32_16x16x32_f16(
                     false, a, false, b0, (short)0, c0, false, false);
            c1 = __builtin_amdgcn_wmma_f32_16x16x32_f16(
                     false, a, false, b1, (short)0, c1, false, false);
        }

#pragma unroll
        for (int r = 0; r < 8; ++r) {
            s_part[wave][0][r][lane] = c0[r];
            s_part[wave][1][r][lane] = c1[r];
        }
    }
    __syncthreads();

    // Reduce partial C tiles across waves; scatter to s_W[token][out_idx].
    for (int idx = tid; idx < 512; idx += 256) {
        const int nt  = idx >> 8;
        const int rem = idx & 255;
        const int r   = rem >> 5;
        const int ln  = rem & 31;
        float s = 0.0f;
#pragma unroll
        for (int w = 0; w < NWAVES; ++w) s += s_part[w][nt][r][ln];
        const int mm = r + ((ln >> 4) << 3);   // C layout: m = r + 8*(lane/16)
        const int nn = nt * 16 + (ln & 15);    //           n = lane%16
        s_W[mm][nn] = s;
    }
    __syncthreads();

    // ---------------- Phase 3: gates + Sinkhorn-Knopp -> mixing matrix M ----
    if (tid < TILE_S) {
        const float a0 = alpha[0], a1 = alpha[1], a2 = alpha[2];
        float hpre[4], hpost[4], h[4][4];
#pragma unroll
        for (int j = 0; j < 4; ++j)
            hpre[j] = sigmoidf_(s_W[tid][j] * a0 + beta[j]) + EPS;
#pragma unroll
        for (int i = 0; i < 4; ++i)
            hpost[i] = 2.0f * sigmoidf_(s_W[tid][4 + i] * a1 + beta[4 + i]);
#pragma unroll
        for (int r = 0; r < 4; ++r)
#pragma unroll
            for (int c = 0; c < 4; ++c)
                h[r][c] = s_W[tid][8 + 4 * r + c] * a2 + beta[8 + 4 * r + c];

        // softmax over rows, + eps
#pragma unroll
        for (int r = 0; r < 4; ++r) {
            float mx = fmaxf(fmaxf(h[r][0], h[r][1]), fmaxf(h[r][2], h[r][3]));
            float e0 = __expf(h[r][0] - mx), e1 = __expf(h[r][1] - mx);
            float e2 = __expf(h[r][2] - mx), e3 = __expf(h[r][3] - mx);
            float inv = 1.0f / (e0 + e1 + e2 + e3);
            h[r][0] = e0 * inv + EPS; h[r][1] = e1 * inv + EPS;
            h[r][2] = e2 * inv + EPS; h[r][3] = e3 * inv + EPS;
        }
        // first column normalization
#pragma unroll
        for (int c = 0; c < 4; ++c) {
            float inv = 1.0f / (h[0][c] + h[1][c] + h[2][c] + h[3][c] + EPS);
#pragma unroll
            for (int r = 0; r < 4; ++r) h[r][c] *= inv;
        }
        // remaining iterations: row then column
        for (int it = 0; it < SINK_ITERS - 1; ++it) {
#pragma unroll
            for (int r = 0; r < 4; ++r) {
                float inv = 1.0f / (h[r][0] + h[r][1] + h[r][2] + h[r][3] + EPS);
#pragma unroll
                for (int c = 0; c < 4; ++c) h[r][c] *= inv;
            }
#pragma unroll
            for (int c = 0; c < 4; ++c) {
                float inv = 1.0f / (h[0][c] + h[1][c] + h[2][c] + h[3][c] + EPS);
#pragma unroll
                for (int r = 0; r < 4; ++r) h[r][c] *= inv;
            }
        }
        // out[i,:] = sum_j (h_post[i]*h_pre[j] + h_res[j,i]) * xr[j,:]
#pragma unroll
        for (int i = 0; i < 4; ++i)
#pragma unroll
            for (int j = 0; j < 4; ++j)
                s_M[tid][i * 4 + j] = hpost[i] * hpre[j] + h[j][i];
    }
    __syncthreads();

    // ---------------- Phase 4: out = M . xr  (float4 vectorized, coalesced) --
    for (int it = 0; it < 16; ++it) {
        const int lin   = tid + it * 256;  // 0..4095 over (token, t4)
        const int token = lin >> 8;        // 16 tokens
        const int t4    = lin & 255;       // 256 float4 per 1024-wide stream
        const float4* xv = (const float4*)(x   + (size_t)(s0 + token) * D_TOT);
        float4*       ov = (float4*)      (out + (size_t)(s0 + token) * D_TOT);
        float4 xj[4];
#pragma unroll
        for (int j = 0; j < 4; ++j) xj[j] = xv[j * 256 + t4];
#pragma unroll
        for (int i = 0; i < 4; ++i) {
            const float m0 = s_M[token][i * 4 + 0];
            const float m1 = s_M[token][i * 4 + 1];
            const float m2 = s_M[token][i * 4 + 2];
            const float m3 = s_M[token][i * 4 + 3];
            float4 o;
            o.x = m0 * xj[0].x + m1 * xj[1].x + m2 * xj[2].x + m3 * xj[3].x;
            o.y = m0 * xj[0].y + m1 * xj[1].y + m2 * xj[2].y + m3 * xj[3].y;
            o.z = m0 * xj[0].z + m1 * xj[1].z + m2 * xj[2].z + m3 * xj[3].z;
            o.w = m0 * xj[0].w + m1 * xj[1].w + m2 * xj[2].w + m3 * xj[3].w;
            ov[i * 256 + t4] = o;
        }
    }
}

extern "C" void kernel_launch(void* const* d_in, const int* in_sizes, int n_in,
                              void* d_out, int out_size, void* d_ws, size_t ws_size,
                              hipStream_t stream) {
    (void)in_sizes; (void)n_in; (void)out_size; (void)d_ws; (void)ws_size;
    const float* x      = (const float*)d_in[0];
    const float* phi    = (const float*)d_in[1];
    const float* alpha  = (const float*)d_in[2];
    const float* beta   = (const float*)d_in[3];
    const float* gamma_ = (const float*)d_in[4];
    float* out = (float*)d_out;

    dim3 grid(S_TOT / TILE_S);  // 256 workgroups
    dim3 block(256);            // 8 wave32 per workgroup
    mhc_fused_kernel<<<grid, block, 0, stream>>>(x, phi, alpha, beta, gamma_, out);
}
